// ChemicalEquationCRF_40699110097399
// MI455X (gfx1250) — matrix operand order repaired and checked by reading, not compiled, not run on hardware
//
#include <hip/hip_runtime.h>
#include <hip/hip_bf16.h>
#include <math.h>

// Problem constants (from reference): B=128, T=1024, V=192
#define BB 128
#define TT 1024
#define VV 192
#define NT 12          // N tiles of 16 (192/16)
#define KT 6           // K tiles of 32 (192/32)
#define STR 200        // padded LDS row stride in halves (400B = 25*16B, conflict-free)
#define EPSC 1e-8f

typedef _Float16 half8 __attribute__((ext_vector_type(8)));
typedef _Float16 v16h  __attribute__((ext_vector_type(16)));
typedef float    v8f   __attribute__((ext_vector_type(8)));

// Fast transcendentals: arguments are known to be in safe ranges
// (log arg >= 1e-8, exp arg <= 0), so skip the denormal/overflow fixups.
__device__ __forceinline__ float fast_log(float x) {
#if __has_builtin(__builtin_amdgcn_logf)
    return __builtin_amdgcn_logf(x) * 0.69314718055994531f;   // v_log_f32 * ln2
#else
    return __logf(x);
#endif
}
__device__ __forceinline__ float fast_exp(float x) {
#if __has_builtin(__builtin_amdgcn_exp2f)
    return __builtin_amdgcn_exp2f(x * 1.44269504088896340f);  // v_exp_f32
#else
    return __expf(x);
#endif
}

// ---------------------------------------------------------------------------
// Pack E = exp(transitions) into WMMA B fragments in this file's contiguous k order:
// flat index ((nt*KT + kt)*32 + lane)*16 + h
//   lane 0-15 : N = nt*16 + lane,      halves h -> K = kt*32 + h       (K 0..15)
//   lane 16-31: N = nt*16 + (lane-16), halves h -> K = kt*32 + 16 + h  (K 16..31)
// ---------------------------------------------------------------------------
__global__ void crf_pack_B(const float* __restrict__ trans, _Float16* __restrict__ Bp) {
    int gid = blockIdx.x * 256 + threadIdx.x;
    if (gid >= NT * KT * 32 * 16) return;
    int h    = gid & 15;
    int lane = (gid >> 4) & 31;
    int kt   = (gid >> 9) % KT;
    int nt   = gid / (16 * 32 * KT);
    int n = nt * 16 + (lane & 15);
    int k = kt * 32 + ((lane >= 16) ? 16 : 0) + h;
    Bp[gid] = (_Float16)expf(trans[k * VV + n]);
}

// ---------------------------------------------------------------------------
// Gold (numerator) score + sequence length per batch. One block per batch.
// ws[b] = gold score, ws[128+b] = clipped sequence length.
// ---------------------------------------------------------------------------
__global__ __launch_bounds__(256) void crf_gold(
    const float* __restrict__ em, const int* __restrict__ tags,
    const int* __restrict__ mask, const float* __restrict__ trans,
    const float* __restrict__ startt, const float* __restrict__ endt,
    float* __restrict__ wsf) {
    int b = blockIdx.x;
    __shared__ float s_sc[256];
    __shared__ int   s_ln[256];
    float acc = 0.0f;
    int   len = 0;
    for (int t = threadIdx.x; t < TT; t += 256) {
        int tg = tags[b * TT + t];
        int mk = mask[b * TT + t];
        len += mk;
        if (t == 0) {
            acc += startt[tg] + em[(size_t)b * TT * VV + tg];
        } else {
            int tp = tags[b * TT + t - 1];
            float c = em[(size_t)b * TT * VV + (size_t)t * VV + tg] + trans[tp * VV + tg];
            acc += c * (float)mk;
        }
    }
    s_sc[threadIdx.x] = acc;
    s_ln[threadIdx.x] = len;
    __syncthreads();
    for (int off = 128; off > 0; off >>= 1) {
        if (threadIdx.x < off) {
            s_sc[threadIdx.x] += s_sc[threadIdx.x + off];
            s_ln[threadIdx.x] += s_ln[threadIdx.x + off];
        }
        __syncthreads();
    }
    if (threadIdx.x == 0) {
        int L = s_ln[0];
        int last = (L > 0) ? (L - 1) : 0;
        wsf[b]       = s_sc[0] + endt[tags[b * TT + last]];
        wsf[BB + b]  = fmaxf((float)L, 1.0f);
    }
}

// ---------------------------------------------------------------------------
// Forward log-partition as a chain of WMMA GEMMs.
// Grid: B/16 workgroups; block: 384 threads = 12 waves (one 16-col N-tile each).
// Each step: rowmax -> p = exp(alpha-m) in LDS -> 6x v_wmma_f32_16x16x32_f16
// per wave (B-fragments of exp(trans) held in VGPRs) -> alpha update.
// ---------------------------------------------------------------------------
__global__ __launch_bounds__(384) void crf_fwd(
    const float* __restrict__ em, const int* __restrict__ mask,
    const float* __restrict__ startt, const float* __restrict__ endt,
    const _Float16* __restrict__ Bp, const float* __restrict__ wsf,
    float* __restrict__ out) {

    __shared__ __align__(16) float      s_alpha[16 * VV];
    __shared__ __align__(16) _Float16   s_p[16 * STR];
    __shared__ float                    s_red[16 * NT];
    __shared__ __align__(16) float      s_m[16];
    __shared__ int                      s_mk[16];

    const int tid  = threadIdx.x;
    const int lane = tid & 31;
    const int w    = tid >> 5;        // wave id == N tile
    const int b0   = blockIdx.x * 16;
    const int row16 = lane & 15;
    const int hi    = lane >> 4;      // 0: lanes 0-15, 1: lanes 16-31
    const int col   = w * 16 + row16; // this lane's output column (state index)

    // p-phase ownership: each thread handles 8 contiguous cols of one row.
    const int pr = tid / 24;          // row 0..15
    const int ps = (tid % 24) * 8;    // col start (0..184 step 8)

    // alpha0 = start + emissions[:,0,:]
    for (int i = tid; i < 16 * VV; i += 384) {
        int r = i / VV, j = i % VV;
        s_alpha[i] = startt[j] + em[((size_t)(b0 + r) * TT) * VV + j];
    }

    // Load constant B fragments (exp(trans)) for this wave's N tile: 48 VGPRs.
    v16h bf[KT];
    const v16h* Bv = (const v16h*)Bp;
#pragma unroll
    for (int kt = 0; kt < KT; ++kt) bf[kt] = Bv[(w * KT + kt) * 32 + lane];

    __syncthreads();

    // Register copy of this lane's 8 owned alpha values (rows hi*8..hi*8+7, col).
    float areg[8];
#pragma unroll
    for (int v = 0; v < 8; ++v) areg[v] = s_alpha[(v + hi * 8) * VV + col];

    // Hoisted pointers, bumped by VV floats per step (no per-step 64-bit mads).
    const float* emcol = em + ((size_t)(b0 + hi * 8) * TT + 1) * VV + col;    // update loads
    const float* pfp   = em + ((size_t)(b0 + (tid / NT)) * TT + 2) * VV + (tid % NT) * 16; // prefetch
    const int*   mkp   = mask + (b0 + tid) * TT + 1;                          // tid<16 only

    for (int t = 1; t < TT; ++t) {
        // Prefetch emissions for step t+1 into cache (global_prefetch_b8).
        if (t + 1 < TT && tid < 192) __builtin_prefetch(pfp, 0, 0);
        // One mask value per row, loaded once per step.
        if (tid < 16) s_mk[tid] = *mkp;

        // Row max partials: 192 threads, one (row, 16-col chunk) each, float4 reads.
        if (tid < 192) {
            int r = tid / NT, ch = tid % NT;
            const float4* a = (const float4*)&s_alpha[r * VV + ch * 16];
            float4 x0 = a[0], x1 = a[1], x2 = a[2], x3 = a[3];
            float mx = fmaxf(fmaxf(fmaxf(x0.x, x0.y), fmaxf(x0.z, x0.w)),
                             fmaxf(fmaxf(x1.x, x1.y), fmaxf(x1.z, x1.w)));
            mx = fmaxf(mx, fmaxf(fmaxf(fmaxf(x2.x, x2.y), fmaxf(x2.z, x2.w)),
                                 fmaxf(fmaxf(x3.x, x3.y), fmaxf(x3.z, x3.w))));
            s_red[r * NT + ch] = mx;
        }
        __syncthreads();
        if (tid < 16) {
            float mx = s_red[tid * NT];
#pragma unroll
            for (int q = 1; q < NT; ++q) mx = fmaxf(mx, s_red[tid * NT + q]);
            s_m[tid] = mx;
        }
        __syncthreads();

        // p = exp(alpha - m): 8 contiguous cols per thread, one ds_store_b128.
        {
            const float* ap = &s_alpha[pr * VV + ps];
            float m = s_m[pr];
            half8 ph;
#pragma unroll
            for (int q = 0; q < 8; ++q) ph[q] = (_Float16)fast_exp(ap[q] - m);
            *(half8*)&s_p[pr * STR + ps] = ph;
        }
        __syncthreads();

        // Issue this step's emission loads NOW so WMMA+ds latency hides them.
        float emv[8];
#pragma unroll
        for (int v = 0; v < 8; ++v) emv[v] = emcol[(size_t)v * TT * VV];
        // Lane's 8 row-maxes from LDS (two b64/b128 reads instead of 8 b32).
        float4 m01 = *(const float4*)&s_m[hi * 8];
        float4 m23 = *(const float4*)&s_m[hi * 8 + 4];
        float mrow[8] = {m01.x, m01.y, m01.z, m01.w, m23.x, m23.y, m23.z, m23.w};

        // GEMM tile: S = p (16x192) @ E-tile (192x16), f32 accumulate.
        v8f c = {0.f, 0.f, 0.f, 0.f, 0.f, 0.f, 0.f, 0.f};
#pragma unroll
        for (int kt = 0; kt < KT; ++kt) {
            // A-fragment, contiguous k order (matches packed B):
            //  lane<16 (M=lane):    halves 0..15 = K kt*32+0..15
            //  lane>=16 (M=lane-16):halves 0..15 = K kt*32+16..31
            int k0 = kt * 32 + hi * 16;
            half8 h0 = *(const half8*)&s_p[row16 * STR + k0];
            half8 h1 = *(const half8*)&s_p[row16 * STR + k0 + 8];
            v16h a = __builtin_shufflevector(h0, h1,
                0, 1, 2, 3, 4, 5, 6, 7, 8, 9, 10, 11, 12, 13, 14, 15);
            c = __builtin_amdgcn_wmma_f32_16x16x32_f16(
                    false, a, false, bf[kt], (short)0, c, false, false);
        }
        __syncthreads();   // all reads of s_p / s_alpha done before updates

        // alpha[r][col] = m[r] + log(S + eps) + emit ; keep old where mask==0.
#pragma unroll
        for (int v = 0; v < 8; ++v) {
            int r = v + hi * 8;
            float na = mrow[v] + fast_log(c[v] + EPSC) + emv[v];
            na = (s_mk[r] != 0) ? na : areg[v];   // branchless keep-old
            areg[v] = na;
            s_alpha[r * VV + col] = na;
        }
        __syncthreads();

        emcol += VV;
        pfp   += VV;
        mkp   += 1;
    }

    // log_den = lse(alpha + end_trans) per row; out = (log_den - gold)/len.
    if (tid < 192) {
        int r = tid / NT, ch = tid % NT;
        float mx = -3.0e38f;
#pragma unroll
        for (int q = 0; q < 16; ++q)
            mx = fmaxf(mx, s_alpha[r * VV + ch * 16 + q] + endt[ch * 16 + q]);
        s_red[r * NT + ch] = mx;
    }
    __syncthreads();
    if (tid < 16) {
        float mx = s_red[tid * NT];
#pragma unroll
        for (int q = 1; q < NT; ++q) mx = fmaxf(mx, s_red[tid * NT + q]);
        s_m[tid] = mx;
    }
    __syncthreads();
    if (tid < 192) {
        int r = tid / NT, ch = tid % NT;
        float s = 0.0f;
#pragma unroll
        for (int q = 0; q < 16; ++q)
            s += expf(s_alpha[r * VV + ch * 16 + q] + endt[ch * 16 + q] - s_m[r]);
        s_red[r * NT + ch] = s;
    }
    __syncthreads();
    if (tid < 16) {
        float s = 0.0f;
#pragma unroll
        for (int q = 0; q < NT; ++q) s += s_red[tid * NT + q];
        float logden = s_m[tid] + logf(s + EPSC);
        int b = b0 + tid;
        out[b] = (logden - wsf[b]) / wsf[BB + b];
    }
}

// ---------------------------------------------------------------------------
extern "C" void kernel_launch(void* const* d_in, const int* in_sizes, int n_in,
                              void* d_out, int out_size, void* d_ws, size_t ws_size,
                              hipStream_t stream) {
    const float* em     = (const float*)d_in[0];   // (B,T,V) f32
    const int*   tags   = (const int*)d_in[1];     // (B,T) int (jax default x32)
    const int*   mask   = (const int*)d_in[2];     // (B,T) int32
    const float* trans  = (const float*)d_in[3];   // (V,V) f32
    const float* startt = (const float*)d_in[4];   // (V,) f32
    const float* endt   = (const float*)d_in[5];   // (V,) f32
    float*       out    = (float*)d_out;           // (B,) f32

    float*     wsf = (float*)d_ws;                 // [0..127] gold, [128..255] len
    _Float16*  Bp  = (_Float16*)(wsf + 2 * BB);    // packed exp(trans) B-fragments

    crf_pack_B<<<(NT * KT * 32 * 16 + 255) / 256, 256, 0, stream>>>(trans, Bp);
    crf_gold<<<BB, 256, 0, stream>>>(em, tags, mask, trans, startt, endt, wsf);
    crf_fwd<<<BB / 16, 384, 0, stream>>>(em, mask, startt, endt, Bp, wsf, out);
}
